// SelectiveAttention_7430293422606
// MI455X (gfx1250) — compile-verified, matrix operand order checked by reading, not run on hardware
//
#include <hip/hip_runtime.h>
#include <hip/hip_bf16.h>
#include <math.h>

// ---------------- problem dims ----------------
static constexpr int kB  = 16;
static constexpr int kS  = 512;
static constexpr int kD  = 1024;
static constexpr int kH  = 16;
static constexpr int kDH = 64;

typedef __attribute__((ext_vector_type(16))) _Float16 v16h;
typedef __attribute__((ext_vector_type(8)))  float    v8f;
typedef __attribute__((ext_vector_type(4)))  float    v4f;   // clang ext-vector: OK for NT builtins

// ---------------- helpers ----------------
static __device__ __forceinline__ float warpReduceSum(float v) {
  #pragma unroll
  for (int off = 16; off > 0; off >>= 1) v += __shfl_down(v, off, 32);
  return v;
}

// Load this lane's 16 f16 fragment values from a contiguous f32 row.
// CDNA5 16-bit A/B layout: per lane the needed K values are two contiguous
// 8-element runs: [base .. base+7] and [base+16 .. base+23], where the caller
// has already added lo = (lane<16 ? 0 : 8) into `base`. 16B-aligned.
static __device__ __forceinline__ v16h frag_from_f32(const float* base) {
  const v4f* p = (const v4f*)base;
  v4f q0 = p[0], q1 = p[1];   // K = +0..+7
  v4f q2 = p[4], q3 = p[5];   // K = +16..+23
  v16h v;
  v[0]=(_Float16)q0.x;  v[1]=(_Float16)q0.y;  v[2]=(_Float16)q0.z;  v[3]=(_Float16)q0.w;
  v[4]=(_Float16)q1.x;  v[5]=(_Float16)q1.y;  v[6]=(_Float16)q1.z;  v[7]=(_Float16)q1.w;
  v[8]=(_Float16)q2.x;  v[9]=(_Float16)q2.y;  v[10]=(_Float16)q2.z; v[11]=(_Float16)q2.w;
  v[12]=(_Float16)q3.x; v[13]=(_Float16)q3.y; v[14]=(_Float16)q3.z; v[15]=(_Float16)q3.w;
  return v;
}

// ---------------- 1a: q0 = Wq x0 + bq ; v0 = Wv x0 + bv ----------------
// one wave per output row; rows = B * 2048
__global__ void k_qv0(const float* __restrict__ x,
                      const float* __restrict__ Wq, const float* __restrict__ bq,
                      const float* __restrict__ Wv, const float* __restrict__ bv,
                      float* __restrict__ q0vec, float* __restrict__ v0) {
  int wave = threadIdx.x >> 5, lane = threadIdx.x & 31;
  int gid  = blockIdx.x * (blockDim.x >> 5) + wave;
  int b = gid >> 11, r2 = gid & 2047;
  const float* xr = x + (size_t)b * kS * kD;     // row 0 of batch b
  const float *W, *bias; float* dst; int r;
  if (r2 < kD) { W = Wq; bias = bq; dst = q0vec + (size_t)b * kD; r = r2; }
  else         { W = Wv; bias = bv; dst = v0    + (size_t)b * kD; r = r2 - kD; }
  const float* w = W + (size_t)r * kD;
  float acc = 0.f;
  for (int k = lane; k < kD; k += 32) acc += w[k] * xr[k];
  acc = warpReduceSum(acc);
  if (lane == 0) dst[r] = acc + bias[r];
}

// ---------------- 1b: u[b,h,:] = sum_dh q0[b,h,dh] * Wk[h*64+dh,:] ----------------
__global__ void k_fold_u(const float* __restrict__ Wk, const float* __restrict__ bk,
                         const float* __restrict__ q0vec,
                         float* __restrict__ u, float* __restrict__ s_bias) {
  int idx = blockIdx.x * blockDim.x + threadIdx.x;   // b*H*D + h*D + d
  int d  = idx & (kD - 1);
  int bh = idx >> 10;
  int h  = bh & (kH - 1);
  int b  = bh >> 4;
  const float* q = q0vec + (size_t)b * kD + h * kDH;
  float acc = 0.f;
  #pragma unroll 8
  for (int t = 0; t < kDH; ++t) acc += q[t] * Wk[(size_t)(h * kDH + t) * kD + d];
  u[(size_t)bh * kD + d] = acc;
  if (d == 0) {
    float sb = 0.f;
    for (int t = 0; t < kDH; ++t) sb += q[t] * bk[h * kDH + t];
    s_bias[bh] = sb;
  }
}

// ---------------- 2: scores[b,h,j] = (u[b,h]·x[b,j] + s_bias)/8  (WMMA) --------------
// M=16(heads) x N=512(j) x K=1024 per batch; one wave per 16x16 tile.
__global__ void k_scores_wmma(const float* __restrict__ x, const float* __restrict__ u,
                              const float* __restrict__ s_bias, float* __restrict__ scores) {
  int wave = threadIdx.x >> 5, lane = threadIdx.x & 31;
  int tile = blockIdx.x * (blockDim.x >> 5) + wave;   // 0 .. B*32-1
  int b  = tile >> 5;
  int jt = tile & 31;
  int m  = lane & 15;
  int lo = (lane < 16) ? 0 : 8;
  const float* arow = u + ((size_t)(b * kH) + m) * kD;                // A row: head m
  const float* brow = x + ((size_t)b * kS + (jt * 16 + m)) * kD;     // B col: j = jt*16+m
  v8f acc = {};
  for (int k0 = 0; k0 < kD; k0 += 32) {
    v16h af = frag_from_f32(arow + k0 + lo);
    v16h bf = frag_from_f32(brow + k0 + lo);
    acc = __builtin_amdgcn_wmma_f32_16x16x32_f16(false, af, false, bf,
                                                 (short)0, acc, false, false);
  }
  int n  = lane & 15;
  int hi = (lane >= 16) ? 8 : 0;
  #pragma unroll
  for (int r = 0; r < 8; ++r) {
    int h = r + hi;
    scores[((size_t)(b * kH + h)) * kS + jt * 16 + n] =
        (acc[r] + s_bias[b * kH + h]) * 0.125f;   // 1/sqrt(64)
  }
}

// ---------------- 3: softmax over j = 1..511 per (b,h) ----------------
__global__ void k_softmax(const float* __restrict__ scores, float* __restrict__ attn) {
  int bh = blockIdx.x;
  const float* s = scores + (size_t)bh * kS;
  float*       a = attn   + (size_t)bh * kS;
  int tid = threadIdx.x;
  int j1 = tid + 1;        // 1..256
  int j2 = tid + 257;      // 257..512 (512 invalid)
  float v1 = s[j1];
  float v2 = (j2 < kS) ? s[j2] : -INFINITY;

  __shared__ float red[8];
  __shared__ float bmax_s, bsum_s;

  float m = fmaxf(v1, v2);
  #pragma unroll
  for (int off = 16; off > 0; off >>= 1) m = fmaxf(m, __shfl_down(m, off, 32));
  if ((tid & 31) == 0) red[tid >> 5] = m;
  __syncthreads();
  if (tid == 0) {
    float t = red[0];
    for (int i = 1; i < 8; ++i) t = fmaxf(t, red[i]);
    bmax_s = t;
  }
  __syncthreads();
  float bmax = bmax_s;
  float e1 = __expf(v1 - bmax);
  float e2 = (j2 < kS) ? __expf(v2 - bmax) : 0.f;
  float sum = e1 + e2;
  #pragma unroll
  for (int off = 16; off > 0; off >>= 1) sum += __shfl_down(sum, off, 32);
  __syncthreads();
  if ((tid & 31) == 0) red[tid >> 5] = sum;
  __syncthreads();
  if (tid == 0) {
    float t = 0.f;
    for (int i = 0; i < 8; ++i) t += red[i];
    bsum_s = t;
  }
  __syncthreads();
  float inv = 1.0f / bsum_s;
  if (tid == 0) a[0] = 0.f;
  a[j1] = e1 * inv;
  if (j2 < kS) a[j2] = e2 * inv;
}

// ---------------- 4: xbar[b,h,:] = attn0[b,h,:] @ X_b  (WMMA) ----------------
// M=16(heads) x N=1024(d) x K=512(j) per batch; one wave per 16x16 tile.
__global__ void k_xbar_wmma(const float* __restrict__ x, const float* __restrict__ attn,
                            float* __restrict__ xbar) {
  int wave = threadIdx.x >> 5, lane = threadIdx.x & 31;
  int tile = blockIdx.x * (blockDim.x >> 5) + wave;   // 0 .. B*64-1
  int b  = tile >> 6;
  int nt = tile & 63;
  int m  = lane & 15;
  int lo = (lane < 16) ? 0 : 8;
  const float* arow = attn + ((size_t)(b * kH) + m) * kS;   // A row: head m, K=j
  int ncol = nt * 16 + m;                                   // B column: feature d
  const float* xb = x + (size_t)b * kS * kD + ncol;
  v8f acc = {};
  for (int k0 = 0; k0 < kS; k0 += 32) {
    v16h af = frag_from_f32(arow + k0 + lo);
    v16h bf;
    #pragma unroll
    for (int e = 0; e < 8; ++e)
      bf[e]     = (_Float16)xb[(size_t)(k0 + lo + e) * kD];
    #pragma unroll
    for (int e = 0; e < 8; ++e)
      bf[8 + e] = (_Float16)xb[(size_t)(k0 + 16 + lo + e) * kD];
    acc = __builtin_amdgcn_wmma_f32_16x16x32_f16(false, af, false, bf,
                                                 (short)0, acc, false, false);
  }
  int n  = lane & 15;
  int hi = (lane >= 16) ? 8 : 0;
  #pragma unroll
  for (int r = 0; r < 8; ++r) {
    int h = r + hi;
    xbar[((size_t)(b * kH + h)) * kD + nt * 16 + n] = acc[r];
  }
}

// ---------------- 5a: ctx[b,d'] = Wv[d',:]·xbar[b, d'/64, :] + bv[d'] ----------------
__global__ void k_ctx(const float* __restrict__ Wv, const float* __restrict__ bv,
                      const float* __restrict__ xbar, float* __restrict__ ctx) {
  int wave = threadIdx.x >> 5, lane = threadIdx.x & 31;
  int gid  = blockIdx.x * (blockDim.x >> 5) + wave;  // < B*1024
  int b = gid >> 10, dp = gid & 1023, h = dp >> 6;
  const float* w  = Wv   + (size_t)dp * kD;
  const float* xb = xbar + ((size_t)(b * kH + h)) * kD;
  float acc = 0.f;
  for (int k = lane; k < kD; k += 32) acc += w[k] * xb[k];
  acc = warpReduceSum(acc);
  if (lane == 0) ctx[(size_t)b * kD + dp] = acc + bv[dp];
}

// ---------------- 5b: row0 = Wo ctx + bo ; y = Wo v0 + bo ----------------
__global__ void k_outrows(const float* __restrict__ Wo, const float* __restrict__ bo,
                          const float* __restrict__ ctx, const float* __restrict__ v0,
                          float* __restrict__ row0, float* __restrict__ y) {
  int wave = threadIdx.x >> 5, lane = threadIdx.x & 31;
  int gid  = blockIdx.x * (blockDim.x >> 5) + wave;  // < B*2048
  int b = gid >> 11, r2 = gid & 2047;
  const float* vec; float* dst; int r;
  if (r2 < kD) { vec = ctx + (size_t)b * kD; dst = row0 + (size_t)b * kD; r = r2; }
  else         { vec = v0  + (size_t)b * kD; dst = y    + (size_t)b * kD; r = r2 - kD; }
  const float* w = Wo + (size_t)r * kD;
  float acc = 0.f;
  for (int k = lane; k < kD; k += 32) acc += w[k] * vec[k];
  acc = warpReduceSum(acc);
  if (lane == 0) dst[r] = acc + bo[r];
}

// ---------------- 6a: fill out (B,S,D): row0 at i==0, y broadcast elsewhere -------
// Write-once streaming output -> nontemporal b128 stores (TH=NT) to keep the
// 192 MB L2 from write-allocating 32 MB of never-re-read data.
__global__ void k_fill_out(const float* __restrict__ row0, const float* __restrict__ y,
                           float* __restrict__ out) {
  size_t i4 = (size_t)blockIdx.x * blockDim.x + threadIdx.x;  // < B*S*D/4 = 2^21
  int b   = (int)(i4 >> 17);          // S*D/4 = 131072
  int rem = (int)(i4 & 131071);
  int i   = rem >> 8;                 // D/4 = 256
  int d4  = rem & 255;
  const v4f* src = (const v4f*)((i == 0 ? row0 : y) + (size_t)b * kD);
  v4f v = src[d4];
  __builtin_nontemporal_store(v, ((v4f*)out) + i4);
}

// ---------------- 6b: fill attn (B,H,S,S) ----------------
// i==0: softmax row (attn0, with [0]=0); i>0: one-hot at j==0.
// 268 MB of write-once output -> nontemporal b128 stores.
__global__ void k_fill_attn(const float* __restrict__ attn0, float* __restrict__ attn_out) {
  size_t i4 = (size_t)blockIdx.x * blockDim.x + threadIdx.x;  // < B*H*S*S/4 = 2^24
  int bh  = (int)(i4 >> 16);          // S*S/4 = 65536
  int rem = (int)(i4 & 65535);
  int i   = rem >> 7;                 // S/4 = 128
  int j4  = rem & 127;
  v4f v;
  if (i == 0) {
    v = ((const v4f*)(attn0 + (size_t)bh * kS))[j4];
  } else {
    v = (v4f){ (j4 == 0) ? 1.0f : 0.0f, 0.f, 0.f, 0.f };
  }
  __builtin_nontemporal_store(v, ((v4f*)attn_out) + i4);
}

// ---------------- host launcher ----------------
extern "C" void kernel_launch(void* const* d_in, const int* in_sizes, int n_in,
                              void* d_out, int out_size, void* d_ws, size_t ws_size,
                              hipStream_t stream) {
  const float* x  = (const float*)d_in[0];
  const float* Wq = (const float*)d_in[1];
  const float* bq = (const float*)d_in[2];
  const float* Wk = (const float*)d_in[3];
  const float* bk = (const float*)d_in[4];
  const float* Wv = (const float*)d_in[5];
  const float* bv = (const float*)d_in[6];
  const float* Wo = (const float*)d_in[7];
  const float* bo = (const float*)d_in[8];

  float* out      = (float*)d_out;                        // (B,S,D)
  float* attn_out = out + (size_t)kB * kS * kD;           // (B,H,S,S)

  // workspace layout (floats); total ~3.4 MB
  float* ws = (float*)d_ws;
  float* q0vec = ws;                         // B*1024
  float* v0    = q0vec + kB * kD;            // B*1024
  float* u     = v0    + kB * kD;            // B*16*1024
  float* sbias = u     + kB * kH * kD;       // B*16
  float* sc    = sbias + kB * kH;            // B*16*512
  float* at0   = sc    + kB * kH * kS;       // B*16*512
  float* xbar  = at0   + kB * kH * kS;       // B*16*1024
  float* ctx   = xbar  + kB * kH * kD;       // B*1024
  float* row0  = ctx   + kB * kD;            // B*1024
  float* ybr   = row0  + kB * kD;            // B*1024

  // 1a: q0, v0              (B*2048 rows, wave/row, 8 waves/block)
  k_qv0<<<4096, 256, 0, stream>>>(x, Wq, bq, Wv, bv, q0vec, v0);
  // 1b: u, s_bias           (B*H*D threads)
  k_fold_u<<<1024, 256, 0, stream>>>(Wk, bk, q0vec, u, sbias);
  // 2: WMMA scores          (B*32 tiles, 4 waves/block)
  k_scores_wmma<<<128, 128, 0, stream>>>(x, u, sbias, sc);
  // 3: softmax              (B*H blocks)
  k_softmax<<<kB * kH, 256, 0, stream>>>(sc, at0);
  // 4: WMMA xbar            (B*64 tiles, 4 waves/block)
  k_xbar_wmma<<<256, 128, 0, stream>>>(x, at0, xbar);
  // 5a: ctx                 (B*1024 rows)
  k_ctx<<<2048, 256, 0, stream>>>(Wv, bv, xbar, ctx);
  // 5b: row0, y             (B*2048 rows)
  k_outrows<<<4096, 256, 0, stream>>>(Wo, bo, ctx, v0, row0, ybr);
  // 6a: stream out          (2^21 float4, NT stores)
  k_fill_out<<<8192, 256, 0, stream>>>(row0, ybr, out);
  // 6b: stream attn         (2^24 float4, NT stores)
  k_fill_attn<<<65536, 256, 0, stream>>>(at0, attn_out);
}
